// BIDAF_62534723829977
// MI455X (gfx1250) — matrix operand order repaired and checked by reading, not compiled, not run on hardware
//
#include <hip/hip_runtime.h>
#include <math.h>

// ---------------------------------------------------------------------------
// BiDAF on MI455X (gfx1250, wave32).
//  * Input-to-hidden projections + attention GEMMs -> v_wmma_f32_16x16x32_f16,
//    register-blocked 2x2 tiles per wave (32x32), 4 waves per block (64x64).
//  * LSTM recurrences -> one 1024-thread workgroup per direction; k-major
//    packed-f16 Whh (coalesced, half the L2 traffic), fp32 state + accumulate
//    (mixed-precision FMA path), prefetch of next step's gate row.
//  * Elementwise / softmax glue in fp32
// ---------------------------------------------------------------------------

typedef _Float16 f16;
typedef _Float16 v16h __attribute__((ext_vector_type(16)));
typedef _Float16 v8h  __attribute__((ext_vector_type(8)));
typedef _Float16 h2   __attribute__((ext_vector_type(2)));
typedef float    v8f  __attribute__((ext_vector_type(8)));

#define TLEN 4096
#define JLEN 256
#define NEG_BIG -3.402823466e38f

__device__ __forceinline__ float sigmoidf_(float x) { return 1.0f / (1.0f + expf(-x)); }

// A fragment: lane(l<16): M=l, K in {0..7, 16..23}; lane(l>=16): M=l-16, K in {8..15, 24..31}
__device__ __forceinline__ v16h load_afrag(const f16* p)
{
    v8h lo = *(const v8h*)(p);
    v8h hi = *(const v8h*)(p + 16);
    return __builtin_shufflevector(lo, hi, 0,1,2,3,4,5,6,7,8,9,10,11,12,13,14,15);
}

// ---------------------------------------------------------------------------
// WMMA GEMM:  C[z][m,n] = sum_k A[m,k] * W[z][n,k]  (+ b0[z][n] + b1[z][n])
// A: f16 row-major [M,K]; W: f16 row-major [N,K]; C: f32 row-major [M,N].
// Block = 128 threads (4 waves), block tile 64x64, wave tile 32x32 (2x2 WMMA).
// K % 32 == 0, M % 64 == 0, N % 64 == 0.
// ---------------------------------------------------------------------------
__global__ __launch_bounds__(128)
void wmma_gemm_f16(const f16* __restrict__ A, const f16* __restrict__ W,
                   float* __restrict__ C,
                   const float* __restrict__ b0, const float* __restrict__ b1,
                   int N, int K,
                   long long wStride, long long cStride, long long bStride)
{
    const int z      = blockIdx.z;
    const int wave   = threadIdx.x >> 5;        // 0..3
    const int l      = threadIdx.x & 31;
    const int lane16 = l & 15;
    const int hi     = l >> 4;                  // 0 or 1

    const int nb = (blockIdx.x << 6) + ((wave & 1) << 5);   // wave col base
    const int mb = (blockIdx.y << 6) + ((wave >> 1) << 5);  // wave row base

    const f16* Wz = W + (size_t)z * wStride;
    float*     Cz = C + (size_t)z * cStride;

    const f16* Ap0 = A  + (size_t)(mb + lane16) * K + hi * 8;
    const f16* Ap1 = Ap0 + (size_t)16 * K;
    // B fragment: lane(l<16): N=l, K in {0..7, 16..23}; lane(l>=16): N=l-16, K in {8..15, 24..31}
    const f16* Wp0 = Wz + (size_t)(nb + lane16) * K + hi * 8;
    const f16* Wp1 = Wp0 + (size_t)16 * K;

    v8f acc00 = {}, acc01 = {}, acc10 = {}, acc11 = {};
    for (int k0 = 0; k0 < K; k0 += 32) {
        v16h a0  = load_afrag(Ap0 + k0);
        v16h a1  = load_afrag(Ap1 + k0);
        v16h bb0 = load_afrag(Wp0 + k0);
        v16h bb1 = load_afrag(Wp1 + k0);
        acc00 = __builtin_amdgcn_wmma_f32_16x16x32_f16(false, a0, false, bb0, (short)0, acc00, false, false);
        acc01 = __builtin_amdgcn_wmma_f32_16x16x32_f16(false, a0, false, bb1, (short)0, acc01, false, false);
        acc10 = __builtin_amdgcn_wmma_f32_16x16x32_f16(false, a1, false, bb0, (short)0, acc10, false, false);
        acc11 = __builtin_amdgcn_wmma_f32_16x16x32_f16(false, a1, false, bb1, (short)0, acc11, false, false);
    }

    const int n0 = nb + lane16, n1 = n0 + 16;
    float bias0 = 0.0f, bias1 = 0.0f;
    if (b0) { bias0 += b0[(size_t)z * bStride + n0]; bias1 += b0[(size_t)z * bStride + n1]; }
    if (b1) { bias0 += b1[(size_t)z * bStride + n0]; bias1 += b1[(size_t)z * bStride + n1]; }

    // C/D layout: VGPR r -> M = tile_mb + (hi?8:0) + r, N = tile_nb + lane16
    const int m0 = mb + hi * 8, m1 = m0 + 16;
    #pragma unroll
    for (int r = 0; r < 8; ++r) {
        Cz[(size_t)(m0 + r) * N + n0] = acc00[r] + bias0;
        Cz[(size_t)(m0 + r) * N + n1] = acc01[r] + bias1;
        Cz[(size_t)(m1 + r) * N + n0] = acc10[r] + bias0;
        Cz[(size_t)(m1 + r) * N + n1] = acc11[r] + bias1;
    }
}

// ---------------------------------------------------------------------------
// Whh transpose + pack: W [2,1024,256] f32 -> WT2 [2,128,1024] of h2 where
// WT2[dir][q][j] = { W[dir][j][2q], W[dir][j][2q+1] }  (k-major, coalesced)
// ---------------------------------------------------------------------------
__global__ __launch_bounds__(256)
void transpose_whh_f16(const float* __restrict__ W, h2* __restrict__ WT2)
{
    int id  = blockIdx.x * 256 + threadIdx.x;   // 0 .. 262143
    int j   = id & 1023;
    int q   = (id >> 10) & 127;
    int dir = id >> 17;
    const float* row = W + ((size_t)dir * 1024 + j) * 256;
    h2 v;
    v.x = (f16)row[2 * q];
    v.y = (f16)row[2 * q + 1];
    WT2[id] = v;
}

// ---------------------------------------------------------------------------
// LSTM recurrence, one block per direction (gridDim.x == 2), 1024 threads.
// precomp[dir][t][0..1023] = Wih@x_t + bih + bhh (fp32, exact).
// WhhT2: [2,128,1024] h2 (k-major packed f16).  Hout: [L, 512].
// ---------------------------------------------------------------------------
__global__ __launch_bounds__(1024)
void lstm_recur(const float* __restrict__ precomp, const h2* __restrict__ WhhT2,
                float* __restrict__ Hout, int L)
{
    const int dir = blockIdx.x;
    const int j   = threadIdx.x;           // 0..1023, gate-row index
    __shared__ __align__(16) float sh[256];
    __shared__ float sc[256];
    __shared__ float sz[1024];
    if (j < 256) { sh[j] = 0.0f; sc[j] = 0.0f; }
    __syncthreads();

    const h2*    WT     = WhhT2 + (size_t)dir * 128 * 1024;
    const float* pcbase = precomp + (size_t)dir * L * 1024;

    for (int s = 0; s < L; ++s) {
        const int t = dir ? (L - 1 - s) : s;
        const float* pc = pcbase + (size_t)t * 1024;
        if (s + 1 < L) {
            const int tn = dir ? (t - 1) : (t + 1);
            __builtin_prefetch(pcbase + (size_t)tn * 1024 + j, 0, 0);
        }
        float acc = pc[j];
        #pragma unroll 4
        for (int q = 0; q < 128; q += 4) {      // k pairs: covers k = 2q .. 2q+7
            float4 ha = *(const float4*)(sh + 2 * q);
            float4 hb = *(const float4*)(sh + 2 * q + 4);
            h2 w0 = WT[(size_t)(q + 0) * 1024 + j];
            h2 w1 = WT[(size_t)(q + 1) * 1024 + j];
            h2 w2 = WT[(size_t)(q + 2) * 1024 + j];
            h2 w3 = WT[(size_t)(q + 3) * 1024 + j];
            acc += (float)w0.x * ha.x + (float)w0.y * ha.y;
            acc += (float)w1.x * ha.z + (float)w1.y * ha.w;
            acc += (float)w2.x * hb.x + (float)w2.y * hb.y;
            acc += (float)w3.x * hb.z + (float)w3.y * hb.w;
        }
        sz[j] = acc;
        __syncthreads();
        if (j < 256) {
            float zi = sz[j], zf = sz[j + 256], zg = sz[j + 512], zo = sz[j + 768];
            float c = sigmoidf_(zf) * sc[j] + sigmoidf_(zi) * tanhf(zg);
            float h = sigmoidf_(zo) * tanhf(c);
            sc[j] = c; sh[j] = h;
            Hout[(size_t)t * 512 + (dir << 8) + j] = h;
        }
        __syncthreads();
    }
}

// --------------------------- small utility kernels --------------------------

__global__ void cast_f32_f16(const float* __restrict__ s, f16* __restrict__ d, long long n)
{
    long long i      = (long long)blockIdx.x * blockDim.x + threadIdx.x;
    long long stride = (long long)gridDim.x * blockDim.x;
    for (; i < n; i += stride) d[i] = (f16)s[i];
}

// d[i] = (f16)(s[i] * w[i & mask])     (mask = 511 -> per-column scale by w_hu)
__global__ void scale_cast_f16(const float* __restrict__ s, const float* __restrict__ w,
                               f16* __restrict__ d, long long n, int mask)
{
    long long i      = (long long)blockIdx.x * blockDim.x + threadIdx.x;
    long long stride = (long long)gridDim.x * blockDim.x;
    for (; i < n; i += stride) d[i] = (f16)(s[i] * w[i & mask]);
}

// Ut[k,j] = U[j,k];  U: [256,512] f32 -> Ut: [512,256] f16. grid 512 x 256thr.
__global__ __launch_bounds__(256)
void transpose_cast_f16(const float* __restrict__ U, f16* __restrict__ Ut)
{
    int id = blockIdx.x * 256 + threadIdx.x;    // 0..131071
    int j  = id & 255;
    int k  = id >> 8;
    Ut[id] = (f16)U[(size_t)j * 512 + k];
}

// out[t] = dot(X[t, 0..C-1], w).  One block (256 thr) per row.
__global__ __launch_bounds__(256)
void rowdot(const float* __restrict__ X, const float* __restrict__ w,
            float* __restrict__ out, int C)
{
    __shared__ float sbuf[256];
    int t = blockIdx.x, j = threadIdx.x;
    float acc = 0.0f;
    for (int k = j; k < C; k += 256) acc += X[(size_t)t * C + k] * w[k];
    sbuf[j] = acc; __syncthreads();
    for (int off = 128; off > 0; off >>= 1) {
        if (j < off) sbuf[j] += sbuf[j + off];
        __syncthreads();
    }
    if (j == 0) out[t] = sbuf[0];
}

// S[t,j] += HW[t] + UW[j] + ws_b; maxS[t] = max_j S; A16[t,j] = softmax_j(S) (f16)
__global__ __launch_bounds__(256)
void s_epilogue_softmax(float* __restrict__ S, const float* __restrict__ HW,
                        const float* __restrict__ UW, const float* __restrict__ wsb,
                        float* __restrict__ maxS, f16* __restrict__ A16)
{
    __shared__ float sbuf[256];
    int t = blockIdx.x, j = threadIdx.x;
    float v = S[(size_t)t * 256 + j] + HW[t] + UW[j] + wsb[0];
    sbuf[j] = v; __syncthreads();
    for (int off = 128; off > 0; off >>= 1) {
        if (j < off) sbuf[j] = fmaxf(sbuf[j], sbuf[j + off]);
        __syncthreads();
    }
    float m = sbuf[0]; __syncthreads();
    if (j == 0) maxS[t] = m;
    float e = expf(v - m);
    sbuf[j] = e; __syncthreads();
    for (int off = 128; off > 0; off >>= 1) {
        if (j < off) sbuf[j] += sbuf[j + off];
        __syncthreads();
    }
    A16[(size_t)t * 256 + j] = (f16)(e / sbuf[0]);
}

// bvec = softmax(maxS) over 4096 entries. Single block, 1024 threads.
__global__ __launch_bounds__(1024)
void bvec_softmax(const float* __restrict__ maxS, float* __restrict__ bvec)
{
    __shared__ float sbuf[1024];
    int t = threadIdx.x;
    float m = NEG_BIG;
    for (int i = t; i < TLEN; i += 1024) m = fmaxf(m, maxS[i]);
    sbuf[t] = m; __syncthreads();
    for (int off = 512; off > 0; off >>= 1) {
        if (t < off) sbuf[t] = fmaxf(sbuf[t], sbuf[t + off]);
        __syncthreads();
    }
    m = sbuf[0]; __syncthreads();
    float s = 0.0f;
    for (int i = t; i < TLEN; i += 1024) s += expf(maxS[i] - m);
    sbuf[t] = s; __syncthreads();
    for (int off = 512; off > 0; off >>= 1) {
        if (t < off) sbuf[t] += sbuf[t + off];
        __syncthreads();
    }
    s = sbuf[0];
    for (int i = t; i < TLEN; i += 1024) bvec[i] = expf(maxS[i] - m) / s;
}

// hh[k] = sum_t bvec[t] * H[t,k]   (k = 0..511; grid 2 x 256thr)
__global__ __launch_bounds__(256)
void q2c_hh(const float* __restrict__ bvec, const float* __restrict__ H,
            float* __restrict__ hh)
{
    int k = blockIdx.x * 256 + threadIdx.x;
    float acc = 0.0f;
    for (int t = 0; t < TLEN; ++t) acc += bvec[t] * H[(size_t)t * 512 + k];
    hh[k] = acc;
}

// G[t] = [H, UU, H*UU, H*hh]   (grid 4096 x 512thr)
__global__ __launch_bounds__(512)
void g_build(const float* __restrict__ H, const float* __restrict__ UU,
             const float* __restrict__ hh, float* __restrict__ G)
{
    int t = blockIdx.x, k = threadIdx.x;
    float h  = H [(size_t)t * 512 + k];
    float uu = UU[(size_t)t * 512 + k];
    float* g = G + (size_t)t * 2048;
    g[k]        = h;
    g[512 + k]  = uu;
    g[1024 + k] = h * uu;
    g[1536 + k] = h * hh[k];
}

// p[t] = dot(G[t], w[0:2048]) + dot(M[t], w[2048:2560]) + wb
__global__ __launch_bounds__(256)
void p_dot(const float* __restrict__ G, const float* __restrict__ M,
           const float* __restrict__ w, const float* __restrict__ wb,
           float* __restrict__ p)
{
    __shared__ float sbuf[256];
    int t = blockIdx.x, j = threadIdx.x;
    float acc = 0.0f;
    const float* g = G + (size_t)t * 2048;
    for (int k = j; k < 2048; k += 256) acc += g[k] * w[k];
    const float* m = M + (size_t)t * 512;
    for (int k = j; k < 512; k += 256) acc += m[k] * w[2048 + k];
    sbuf[j] = acc; __syncthreads();
    for (int off = 128; off > 0; off >>= 1) {
        if (j < off) sbuf[j] += sbuf[j + off];
        __syncthreads();
    }
    if (j == 0) p[t] = sbuf[0] + wb[0];
}

// out[0,:] = log_softmax(p1), out[1,:] = log_softmax(p2). Single block.
__global__ __launch_bounds__(1024)
void logsoftmax2(const float* __restrict__ p1, const float* __restrict__ p2,
                 float* __restrict__ out)
{
    __shared__ float sbuf[1024];
    int t = threadIdx.x;
    for (int r = 0; r < 2; ++r) {
        const float* p = r ? p2 : p1;
        float m = NEG_BIG;
        for (int i = t; i < TLEN; i += 1024) m = fmaxf(m, p[i]);
        sbuf[t] = m; __syncthreads();
        for (int off = 512; off > 0; off >>= 1) {
            if (t < off) sbuf[t] = fmaxf(sbuf[t], sbuf[t + off]);
            __syncthreads();
        }
        m = sbuf[0]; __syncthreads();
        float s = 0.0f;
        for (int i = t; i < TLEN; i += 1024) s += expf(p[i] - m);
        sbuf[t] = s; __syncthreads();
        for (int off = 512; off > 0; off >>= 1) {
            if (t < off) sbuf[t] += sbuf[t + off];
            __syncthreads();
        }
        float lg = logf(sbuf[0]);
        __syncthreads();
        for (int i = t; i < TLEN; i += 1024) out[r * TLEN + i] = p[i] - m - lg;
        __syncthreads();
    }
}

// --------------------------- workspace layout (bytes) -----------------------
static constexpr size_t OFF_H    = 0;                                   // [4096,512] f32
static constexpr size_t OFF_U    = OFF_H    + 4096ull*512*4;            // [256,512]  f32
static constexpr size_t OFF_G    = OFF_U    + 256ull*512*4;             // [4096,2048]f32
static constexpr size_t OFF_M0   = OFF_G    + 4096ull*2048*4;           // [4096,512]
static constexpr size_t OFF_M    = OFF_M0   + 4096ull*512*4;
static constexpr size_t OFF_M2   = OFF_M    + 4096ull*512*4;
static constexpr size_t OFF_S    = OFF_M2   + 4096ull*512*4;            // [4096,256]
static constexpr size_t OFF_UU   = OFF_S    + 4096ull*256*4;            // [4096,512]
static constexpr size_t OFF_PRE  = OFF_UU   + 4096ull*512*4;            // [2,4096,1024] f32
static constexpr size_t OFF_A16  = OFF_PRE  + 2ull*4096*1024*4;         // [4096,2048] f16
static constexpr size_t OFF_B16  = OFF_A16  + 4096ull*2048*2;           // [2,1024,2048] f16
static constexpr size_t OFF_UT16 = OFF_B16  + 2ull*1024*2048*2;         // [512,256] f16
static constexpr size_t OFF_HW   = OFF_UT16 + 512ull*256*2;             // [4096]
static constexpr size_t OFF_UW   = OFF_HW   + 4096ull*4;                // [256]
static constexpr size_t OFF_MAXS = OFF_UW   + 256ull*4;                 // [4096]
static constexpr size_t OFF_BV   = OFF_MAXS + 4096ull*4;                // [4096]
static constexpr size_t OFF_HH   = OFF_BV   + 4096ull*4;                // [512]
static constexpr size_t OFF_P1   = OFF_HH   + 512ull*4;                 // [4096]
static constexpr size_t OFF_P2   = OFF_P1   + 4096ull*4;                // [4096]
static constexpr size_t OFF_WHT  = OFF_P2   + 4096ull*4;                // [2,128,1024] h2

extern "C" void kernel_launch(void* const* d_in, const int* in_sizes, int n_in,
                              void* d_out, int out_size, void* d_ws, size_t ws_size,
                              hipStream_t stream)
{
    (void)in_sizes; (void)n_in; (void)out_size; (void)ws_size;

    const float* ctx_vec  = (const float*)d_in[0];
    const float* qry_vec  = (const float*)d_in[1];
    const float* ctx_Wih  = (const float*)d_in[2];
    const float* ctx_Whh  = (const float*)d_in[3];
    const float* ctx_bih  = (const float*)d_in[4];
    const float* ctx_bhh  = (const float*)d_in[5];
    const float* qry_Wih  = (const float*)d_in[6];
    const float* qry_Whh  = (const float*)d_in[7];
    const float* qry_bih  = (const float*)d_in[8];
    const float* qry_bhh  = (const float*)d_in[9];
    const float* mod0_Wih = (const float*)d_in[10];
    const float* mod0_Whh = (const float*)d_in[11];
    const float* mod0_bih = (const float*)d_in[12];
    const float* mod0_bhh = (const float*)d_in[13];
    const float* mod1_Wih = (const float*)d_in[14];
    const float* mod1_Whh = (const float*)d_in[15];
    const float* mod1_bih = (const float*)d_in[16];
    const float* mod1_bhh = (const float*)d_in[17];
    const float* out_Wih  = (const float*)d_in[18];
    const float* out_Whh  = (const float*)d_in[19];
    const float* out_bih  = (const float*)d_in[20];
    const float* out_bhh  = (const float*)d_in[21];
    const float* ws_w     = (const float*)d_in[22];   // [1536] = [w_h | w_u | w_hu]
    const float* ws_b     = (const float*)d_in[23];
    const float* wp1_w    = (const float*)d_in[24];
    const float* wp1_b    = (const float*)d_in[25];
    const float* wp2_w    = (const float*)d_in[26];
    const float* wp2_b    = (const float*)d_in[27];

    char* ws = (char*)d_ws;
    float* H    = (float*)(ws + OFF_H);
    float* U    = (float*)(ws + OFF_U);
    float* G    = (float*)(ws + OFF_G);
    float* M0b  = (float*)(ws + OFF_M0);
    float* Mb   = (float*)(ws + OFF_M);
    float* M2b  = (float*)(ws + OFF_M2);
    float* Sb   = (float*)(ws + OFF_S);
    float* UUb  = (float*)(ws + OFF_UU);
    float* PRE  = (float*)(ws + OFF_PRE);
    f16*   A16  = (f16*)  (ws + OFF_A16);
    f16*   B16  = (f16*)  (ws + OFF_B16);
    f16*   UT16 = (f16*)  (ws + OFF_UT16);
    float* HW   = (float*)(ws + OFF_HW);
    float* UW   = (float*)(ws + OFF_UW);
    float* MAXS = (float*)(ws + OFF_MAXS);
    float* BV   = (float*)(ws + OFF_BV);
    float* HHv  = (float*)(ws + OFF_HH);
    float* P1   = (float*)(ws + OFF_P1);
    float* P2   = (float*)(ws + OFF_P2);
    h2*    WHT  = (h2*)   (ws + OFF_WHT);
    float* OUT  = (float*)d_out;

    auto cast = [&](const float* s, f16* d, long long n) {
        int blocks = (int)((n + 255) / 256);
        cast_f32_f16<<<blocks, 256, 0, stream>>>(s, d, n);
    };
    // C[z] = A @ W[z]^T (+biases); M rows, N cols, K inner; 64x64 block tiles
    auto gemm = [&](const f16* A, const f16* W, float* C,
                    const float* b0, const float* b1,
                    int M, int N, int K, int ndir,
                    long long wStr, long long cStr) {
        dim3 grid(N / 64, M / 64, ndir);
        wmma_gemm_f16<<<grid, 128, 0, stream>>>(A, W, C, b0, b1, N, K,
                                                wStr, cStr, 1024);
    };
    auto recur = [&](const float* Whh, float* Hout, int L) {
        transpose_whh_f16<<<1024, 256, 0, stream>>>(Whh, WHT);
        lstm_recur<<<2, 1024, 0, stream>>>(PRE, WHT, Hout, L);
    };

    // ---- ctx BiLSTM: H = BiLSTM(ctx_vec) -----------------------------------
    cast(ctx_vec, A16, 4096ll * 256);
    cast(ctx_Wih, B16, 2ll * 1024 * 256);
    gemm(A16, B16, PRE, ctx_bih, ctx_bhh, TLEN, 1024, 256, 2,
         1024ll * 256, (long long)TLEN * 1024);
    recur(ctx_Whh, H, TLEN);

    // ---- qry BiLSTM: U = BiLSTM(qry_vec) -----------------------------------
    cast(qry_vec, A16, 256ll * 256);
    cast(qry_Wih, B16, 2ll * 1024 * 256);
    gemm(A16, B16, PRE, qry_bih, qry_bhh, JLEN, 1024, 256, 2,
         1024ll * 256, (long long)JLEN * 1024);
    recur(qry_Whh, U, JLEN);

    // ---- attention ----------------------------------------------------------
    rowdot<<<TLEN, 256, 0, stream>>>(H, ws_w, HW, 512);            // H @ w_h
    rowdot<<<JLEN, 256, 0, stream>>>(U, ws_w + 512, UW, 512);      // U @ w_u
    scale_cast_f16<<<8192, 256, 0, stream>>>(H, ws_w + 1024, A16,
                                             4096ll * 512, 511);   // (H*w_hu) f16
    transpose_cast_f16<<<512, 256, 0, stream>>>(U, UT16);          // U^T f16
    gemm(A16, UT16, Sb, nullptr, nullptr, TLEN, 256, 512, 1, 0, 0); // (H*w_hu)@U^T
    s_epilogue_softmax<<<TLEN, 256, 0, stream>>>(Sb, HW, UW, ws_b, MAXS, A16);
    gemm(A16, UT16, UUb, nullptr, nullptr, TLEN, 512, 256, 1, 0, 0); // A @ U
    bvec_softmax<<<1, 1024, 0, stream>>>(MAXS, BV);
    q2c_hh<<<2, 256, 0, stream>>>(BV, H, HHv);
    g_build<<<TLEN, 512, 0, stream>>>(H, UUb, HHv, G);

    // ---- mod0 BiLSTM (in 2048) ---------------------------------------------
    cast(G, A16, 4096ll * 2048);
    cast(mod0_Wih, B16, 2ll * 1024 * 2048);
    gemm(A16, B16, PRE, mod0_bih, mod0_bhh, TLEN, 1024, 2048, 2,
         1024ll * 2048, (long long)TLEN * 1024);
    recur(mod0_Whh, M0b, TLEN);

    // ---- mod1 BiLSTM (in 512) ----------------------------------------------
    cast(M0b, A16, 4096ll * 512);
    cast(mod1_Wih, B16, 2ll * 1024 * 512);
    gemm(A16, B16, PRE, mod1_bih, mod1_bhh, TLEN, 1024, 512, 2,
         1024ll * 512, (long long)TLEN * 1024);
    recur(mod1_Whh, Mb, TLEN);

    // ---- p1 -----------------------------------------------------------------
    p_dot<<<TLEN, 256, 0, stream>>>(G, Mb, wp1_w, wp1_b, P1);

    // ---- out BiLSTM (in 512) -----------------------------------------------
    cast(Mb, A16, 4096ll * 512);
    cast(out_Wih, B16, 2ll * 1024 * 512);
    gemm(A16, B16, PRE, out_bih, out_bhh, TLEN, 1024, 512, 2,
         1024ll * 512, (long long)TLEN * 1024);
    recur(out_Whh, M2b, TLEN);

    // ---- p2 + log-softmax outputs ------------------------------------------
    p_dot<<<TLEN, 256, 0, stream>>>(G, M2b, wp2_w, wp2_b, P2);
    logsoftmax2<<<1, 1024, 0, stream>>>(P1, P2, OUT);
}
